// QLSTMEnhancedQML_65481071397607
// MI455X (gfx1250) — compile-verified, matrix operand order checked by reading, not run on hardware
//
#include <hip/hip_runtime.h>

// ---------------------------------------------------------------------------
// QLSTM (4-wire) for MI455X / gfx1250.
//   K0 prep : pack Wx -> Bpack2[512][16] float2 (K-pairs), bias16, th16, Wh
//   K1 gemm : pre[t][c] = emb[sent[t]] . Wx_c + b_c  via V_WMMA_F32_16X16X4_F32
//   K2 scan : serial LSTM, closed-form qgate (products of cosines),
//             v_readlane broadcasts + rcp/tanh fast transcendentals
//   K3 head : logits (4->47) + log_softmax
// ---------------------------------------------------------------------------

typedef float v2f __attribute__((ext_vector_type(2)));
typedef float v8f __attribute__((ext_vector_type(8)));
typedef _Float16 v16h __attribute__((ext_vector_type(16)));

#define T_LEN 4096
#define E_DIM 1024
#define DIN   1028
#define NTAG  47

// ---------------- K0: pack weights ----------------------------------------
__global__ __launch_bounds__(256) void qlstm_prep(
    const float* __restrict__ Wf, const float* __restrict__ Wi,
    const float* __restrict__ Wu, const float* __restrict__ Wo,
    const float* __restrict__ bf, const float* __restrict__ bi,
    const float* __restrict__ bu, const float* __restrict__ bo,
    const float* __restrict__ thf, const float* __restrict__ thi,
    const float* __restrict__ thu, const float* __restrict__ tho,
    float* __restrict__ Bpack2, float* __restrict__ bias16,
    float* __restrict__ th16, float* __restrict__ Whpack)
{
  int tid = blockIdx.x * blockDim.x + threadIdx.x;   // 0 .. 16383
  int c = tid & 15;          // column = gate*4 + wire
  int e = tid >> 4;          // 0 .. 1023
  int g = c >> 2, w = c & 3;
  const float* W = (g == 0) ? Wf : (g == 1) ? Wi : (g == 2) ? Wu : Wo;
  // float2 element (e>>1, c): components at float offset (e>>1)*32 + 2c + (e&1)
  Bpack2[(e >> 1) * 32 + c * 2 + (e & 1)] = W[w * DIN + e];
  if (tid < 16) {
    const float* b  = (g == 0) ? bf  : (g == 1) ? bi  : (g == 2) ? bu  : bo;
    const float* th = (g == 0) ? thf : (g == 1) ? thi : (g == 2) ? thu : tho;
    bias16[c] = b[w];
    th16[c]   = th[w];
  }
  if (tid < 64) {
    int cc = tid >> 2, j = tid & 3;
    int gg = cc >> 2, ww = cc & 3;
    const float* W2 = (gg == 0) ? Wf : (gg == 1) ? Wi : (gg == 2) ? Wu : Wo;
    Whpack[cc * 4 + j] = W2[ww * DIN + E_DIM + j];   // recurrent 4x4 per gate
  }
}

// ---------------- K1: gathered GEMM via WMMA -------------------------------
// One wave per 16-token tile; output tile is 16 (tokens) x 16 (gate,wire).
__global__ __launch_bounds__(32) void qlstm_gemm(
    const int*   __restrict__ sent, const float* __restrict__ emb,
    const float* __restrict__ Bpack2, const float* __restrict__ bias16,
    float* __restrict__ pre)
{
  const int tile = blockIdx.x;          // 0 .. 255
  const int lane = threadIdx.x;         // 0 .. 31
  const int hi   = lane >> 4;           // half-wave selector
  const int col  = lane & 15;           // A row (M) and B column (N)

  const int tok = sent[tile * 16 + col];
  const float* __restrict__ arow = emb + (size_t)tok * E_DIM;
  const v2f*   __restrict__ bmat = (const v2f*)Bpack2;   // [512][16] K-pairs

  v8f acc = {};

#if __has_builtin(__builtin_amdgcn_wmma_f32_16x16x4_f32)
  // fp32 path (exact).
  // A frag (16x4): lane holds A[row][k0+2*hi], A[row][k0+2*hi+1]  -> one b64
  // B frag (4x16): lane holds B[k0+2*hi][col], B[k0+2*hi+1][col]  -> one b64
  #pragma unroll 8
  for (int k0 = 0; k0 < E_DIM; k0 += 4) {
    v2f a = *(const v2f*)(arow + k0 + 2 * hi);
    v2f b = bmat[((k0 >> 1) + hi) * 16 + col];
    acc = __builtin_amdgcn_wmma_f32_16x16x4_f32(
        false, a, false, b, (short)0, acc, false, false);
  }
#else
  // fallback: f16 inputs, f32 accumulate (codegen-confirmed builtin)
  for (int k0 = 0; k0 < E_DIM; k0 += 32) {
    v16h a, b;
    #pragma unroll
    for (int j = 0; j < 8; ++j) {
      int kA = k0 + 8 * hi + j;
      int kB = k0 + 16 + 8 * hi + j;
      a[j]     = (_Float16)arow[kA];
      a[j + 8] = (_Float16)arow[kB];
      b[j]     = (_Float16)Bpack2[(kA >> 1) * 32 + col * 2 + (kA & 1)];
      b[j + 8] = (_Float16)Bpack2[(kB >> 1) * 32 + col * 2 + (kB & 1)];
    }
    acc = __builtin_amdgcn_wmma_f32_16x16x32_f16(
        false, a, false, b, (short)0, acc, false, false);
  }
#endif

  const float bv = bias16[col];
  #pragma unroll
  for (int r = 0; r < 8; ++r) {
    int m = r + 8 * hi;                 // C/D layout: M = r (+8 for upper half)
    pre[(tile * 16 + m) * 16 + col] = acc[r] + bv;
  }
}

// ---------------- K2: serial scan with closed-form qgate -------------------
__device__ __forceinline__ float bcast_lane(float x, int lane) {
  int r = __builtin_amdgcn_readlane(__builtin_bit_cast(int, x), lane);
  return __builtin_bit_cast(float, r);
}
__device__ __forceinline__ float sigmoid_f(float x) {
  return __builtin_amdgcn_rcpf(1.0f + __expf(-x));
}
__device__ __forceinline__ float tanh_f(float x) {
#if __has_builtin(__builtin_amdgcn_tanhf)
  return __builtin_amdgcn_tanhf(x);
#elif __has_builtin(__builtin_amdgcn_tanh_f32)
  return __builtin_amdgcn_tanh_f32(x);
#else
  float a = __expf(-2.0f * fabsf(x));
  float t = (1.0f - a) * __builtin_amdgcn_rcpf(1.0f + a);
  return (x >= 0.0f) ? t : -t;
#endif
}

__global__ __launch_bounds__(32) void qlstm_scan(
    const float* __restrict__ pre, const float* __restrict__ Whpack,
    const float* __restrict__ th16, float* __restrict__ hseq)
{
  const int lane = threadIdx.x;
  const int col  = lane & 15;                 // lanes 16-31 mirror 0-15
  const float w0 = Whpack[col * 4 + 0];
  const float w1 = Whpack[col * 4 + 1];
  const float w2 = Whpack[col * 4 + 2];
  const float w3 = Whpack[col * 4 + 3];
  const float tv = th16[col];

  float h0 = 0.f, h1 = 0.f, h2 = 0.f, h3 = 0.f;
  float cw0 = 0.f, cw1 = 0.f, cw2 = 0.f, cw3 = 0.f;

  float pcur = pre[col];
  for (int t = 0; t < T_LEN; ++t) {
    int tn = (t + 1 < T_LEN) ? t + 1 : t;
    float pnext = pre[tn * 16 + col];        // prefetch next step's x-part

    float ang = pcur + tv + w0 * h0 + w1 * h1 + w2 * h2 + w3 * h3;
    float C = __cosf(ang);                   // cos(angle + theta)

    float Cv[16];
    #pragma unroll
    for (int k = 0; k < 16; ++k) Cv[k] = bcast_lane(C, k);  // v_readlane

    // exps per gate: e0=C1C2C3, e1=C0C1, e2=C0C1C2, e3=C0C1C2C3
    float ex[16];
    #pragma unroll
    for (int g = 0; g < 4; ++g) {
      float c0 = Cv[4 * g + 0], c1 = Cv[4 * g + 1];
      float c2 = Cv[4 * g + 2], c3 = Cv[4 * g + 3];
      float t12 = c1 * c2;
      float e1 = c0 * c1;
      float e2 = e1 * c2;
      float e3 = e2 * c3;
      float e0 = t12 * c3;
      ex[4 * g + 0] = e0; ex[4 * g + 1] = e1;
      ex[4 * g + 2] = e2; ex[4 * g + 3] = e3;
    }

    float fw[4], iw[4], uw[4], ow[4];
    #pragma unroll
    for (int w = 0; w < 4; ++w) {
      fw[w] = sigmoid_f(ex[w]);
      iw[w] = sigmoid_f(ex[4 + w]);
      uw[w] = tanh_f  (ex[8 + w]);
      ow[w] = sigmoid_f(ex[12 + w]);
    }
    cw0 = fw[0] * cw0 + iw[0] * uw[0];  h0 = ow[0] * tanh_f(cw0);
    cw1 = fw[1] * cw1 + iw[1] * uw[1];  h1 = ow[1] * tanh_f(cw1);
    cw2 = fw[2] * cw2 + iw[2] * uw[2];  h2 = ow[2] * tanh_f(cw2);
    cw3 = fw[3] * cw3 + iw[3] * uw[3];  h3 = ow[3] * tanh_f(cw3);

    if (lane < 4) {
      float hv = (lane == 0) ? h0 : (lane == 1) ? h1 : (lane == 2) ? h2 : h3;
      hseq[t * 4 + lane] = hv;
    }
    pcur = pnext;
  }
}

// ---------------- K3: tag head + log_softmax -------------------------------
__global__ __launch_bounds__(256) void qlstm_head(
    const float* __restrict__ hseq, const float* __restrict__ Wtag,
    const float* __restrict__ btag, float* __restrict__ out)
{
  int t = blockIdx.x * blockDim.x + threadIdx.x;
  if (t >= T_LEN) return;
  float h0 = hseq[t * 4 + 0], h1 = hseq[t * 4 + 1];
  float h2 = hseq[t * 4 + 2], h3 = hseq[t * 4 + 3];

  float mx = -3.4e38f;
  for (int k = 0; k < NTAG; ++k) {
    float l = btag[k] + Wtag[k * 4 + 0] * h0 + Wtag[k * 4 + 1] * h1
                      + Wtag[k * 4 + 2] * h2 + Wtag[k * 4 + 3] * h3;
    mx = fmaxf(mx, l);
  }
  float s = 0.f;
  for (int k = 0; k < NTAG; ++k) {
    float l = btag[k] + Wtag[k * 4 + 0] * h0 + Wtag[k * 4 + 1] * h1
                      + Wtag[k * 4 + 2] * h2 + Wtag[k * 4 + 3] * h3;
    s += __expf(l - mx);
  }
  float lse = mx + __logf(s);
  for (int k = 0; k < NTAG; ++k) {
    float l = btag[k] + Wtag[k * 4 + 0] * h0 + Wtag[k * 4 + 1] * h1
                      + Wtag[k * 4 + 2] * h2 + Wtag[k * 4 + 3] * h3;
    out[t * NTAG + k] = l - lse;
  }
}

// ---------------- launcher -------------------------------------------------
extern "C" void kernel_launch(void* const* d_in, const int* in_sizes, int n_in,
                              void* d_out, int out_size, void* d_ws, size_t ws_size,
                              hipStream_t stream) {
  (void)in_sizes; (void)n_in; (void)out_size; (void)ws_size;
  const int*   sent = (const int*)  d_in[0];
  const float* emb  = (const float*)d_in[1];
  const float* Wf   = (const float*)d_in[2];
  const float* bf   = (const float*)d_in[3];
  const float* Wi   = (const float*)d_in[4];
  const float* bi   = (const float*)d_in[5];
  const float* Wu   = (const float*)d_in[6];
  const float* bu   = (const float*)d_in[7];
  const float* Wo   = (const float*)d_in[8];
  const float* bo   = (const float*)d_in[9];
  const float* thf  = (const float*)d_in[10];
  const float* thi  = (const float*)d_in[11];
  const float* thu  = (const float*)d_in[12];
  const float* tho  = (const float*)d_in[13];
  const float* Wtag = (const float*)d_in[14];
  const float* btag = (const float*)d_in[15];
  float* out = (float*)d_out;

  float* ws     = (float*)d_ws;
  float* Bpack2 = ws;             // 1024*16 = 16384 floats (as [512][16] float2)
  float* bias16 = ws + 16384;     // 16
  float* th16   = ws + 16400;     // 16
  float* Whpack = ws + 16416;     // 64
  float* pre    = ws + 16480;     // 4096*16 = 65536
  float* hseq   = ws + 82016;     // 4096*4  = 16384

  qlstm_prep<<<64, 256, 0, stream>>>(Wf, Wi, Wu, Wo, bf, bi, bu, bo,
                                     thf, thi, thu, tho,
                                     Bpack2, bias16, th16, Whpack);
  qlstm_gemm<<<T_LEN / 16, 32, 0, stream>>>(sent, emb, Bpack2, bias16, pre);
  qlstm_scan<<<1, 32, 0, stream>>>(pre, Whpack, th16, hseq);
  qlstm_head<<<(T_LEN + 255) / 256, 256, 0, stream>>>(hseq, Wtag, btag, out);
}
